// PredictModel_44418551775738
// MI455X (gfx1250) — compile-verified, matrix operand order checked by reading, not run on hardware
//
#include <hip/hip_runtime.h>
#include <hip/hip_bf16.h>

#define B_DIM 4096
#define P_DIM 64
#define SPOT  512
#define PROT  1280
#define HID   256

typedef __attribute__((ext_vector_type(16))) __bf16 v16bf;
typedef __attribute__((ext_vector_type(8)))  __bf16 v8bf;
typedef __attribute__((ext_vector_type(8)))  float  v8f;
typedef __attribute__((ext_vector_type(4)))  float  v4f;
typedef __attribute__((ext_vector_type(4)))  unsigned int u32x4;
typedef __attribute__((ext_vector_type(8)))  int i32x8;
typedef __attribute__((ext_vector_type(4)))  int i32x4;

#if defined(__AMDGCN__) && __has_builtin(__builtin_amdgcn_tensor_load_to_lds)
#define HAVE_TDM 1
#else
#define HAVE_TDM 0
#endif

// ---------------- fragment helpers -----------------------------------------
// 16-bit A/B fragment K layout (ISA 7.12.2): lanes 0-15 hold K={0..7,16..23},
// lanes 16-31 hold K={8..15,24..31}; A rows / B cols = lane&15.

__device__ __forceinline__ v16bf make_frag(v8bf lo, v8bf hi) {
  v16bf f;
#pragma unroll
  for (int i = 0; i < 8; ++i) { f[i] = lo[i]; f[i + 8] = hi[i]; }
  return f;
}

__device__ __forceinline__ v16bf frag16(const __bf16* p) {
  return make_frag(*(const v8bf*)p, *(const v8bf*)(p + 16));
}

__device__ __forceinline__ v8bf cvt8(v4f x0, v4f x1) {
  v8bf r;
#pragma unroll
  for (int i = 0; i < 4; ++i) { r[i] = (__bf16)x0[i]; r[4 + i] = (__bf16)x1[i]; }
  return r;
}

__device__ __forceinline__ v8f wmma_bf16(v16bf a, v16bf b, v8f c) {
  return __builtin_amdgcn_wmma_f32_16x16x32_bf16(false, a, false, b,
                                                 (short)0, c, false, false);
}

#if HAVE_TDM
// One TDM op: copy a 32(k) x 512(n) bf16 tile of cWt (row stride 1280 elems)
// into LDS at byte offset lds_off, rows packed 64B apart. ISA 8.3/8.4 D#.
__device__ __forceinline__ void tdm_load_b_slab(const __bf16* gsrc,
                                                unsigned lds_off) {
  unsigned long long ga = (unsigned long long)gsrc;
  u32x4 g0;
  g0[0] = 1u;                                   // count=1, user descriptor
  g0[1] = lds_off;                              // LDS byte address
  g0[2] = (unsigned)ga;                         // global addr [31:0]
  g0[3] = (unsigned)((ga >> 32) & 0x01FFFFFFu)  // global addr [56:32]
          | 0x80000000u;                        // type = 2 ("image")
  i32x8 g1;
  g1[0] = 1 << 16;                 // data_size = 2 bytes
  g1[1] = (int)(1280u << 16);      // tensor_dim0 = 1280 (bits [79:48] lo half)
  g1[2] = (int)(512u << 16);       // tensor_dim1 = 512  (bits [111:80] lo half)
  g1[3] = (int)(32u << 16);        // tile_dim0 = 32     (bits [127:112])
  g1[4] = 512;                     // tile_dim1 = 512    (bits [143:128])
  g1[5] = 1280;                    // tensor_dim0_stride (bits [191:160])
  g1[6] = 0;
  g1[7] = 0;
  i32x4 z4 = {0, 0, 0, 0};
#if __clang_major__ >= 23
  i32x8 z8 = {0, 0, 0, 0, 0, 0, 0, 0};
  __builtin_amdgcn_tensor_load_to_lds(g0, g1, z4, z4, z8, 0);
#else
  __builtin_amdgcn_tensor_load_to_lds(g0, g1, z4, z4, 0);
#endif
}
#endif

// ---------------- prep kernels (tiny, one-time weight transforms) -----------

__global__ void k_combined(const float* __restrict__ projW,
                           const float* __restrict__ transW,
                           __bf16* __restrict__ cWt) {
  int id = blockIdx.x * blockDim.x + threadIdx.x;
  if (id >= SPOT * PROT) return;
  int n = id / PROT;
  int d = id % PROT;
  float acc = 0.f;
  for (int m = 0; m < SPOT; ++m)
    acc += projW[d * SPOT + m] * transW[m * SPOT + n];
  cWt[(long)n * PROT + d] = (__bf16)acc;
}

__global__ void k_biasterm(const float* __restrict__ projb,
                           const float* __restrict__ transW,
                           const float* __restrict__ transb,
                           float* __restrict__ biasterm) {
  int n = blockIdx.x * blockDim.x + threadIdx.x;
  if (n >= SPOT) return;
  float acc = transb[n];
  for (int m = 0; m < SPOT; ++m) acc += projb[m] * transW[m * SPOT + n];
  biasterm[n] = acc;
}

__global__ void k_tWbt(const float* __restrict__ transW,
                       __bf16* __restrict__ tWbt) {
  int id = blockIdx.x * blockDim.x + threadIdx.x;
  if (id >= SPOT * SPOT) return;
  int m = id % SPOT, n = id / SPOT;
  tWbt[id] = (__bf16)transW[(SPOT + m) * SPOT + n];
}

__global__ void k_W1t(const float* __restrict__ W1,
                      __bf16* __restrict__ W1t) {
  int id = blockIdx.x * blockDim.x + threadIdx.x;
  if (id >= P_DIM * HID * SPOT) return;
  int s = id % SPOT;
  int h = (id / SPOT) % HID;
  int p = id / (SPOT * HID);
  W1t[id] = (__bf16)W1[((long)p * SPOT + s) * HID + h];
}

// ---------------- cellterm = cell_emb @ Wt_bot + biasterm  (WMMA) -----------
__global__ __launch_bounds__(512) void k_cellterm(
    const float* __restrict__ cell, const __bf16* __restrict__ tWbt,
    const float* __restrict__ biasterm, float* __restrict__ cellterm) {
  __shared__ __align__(16) __bf16 lA[64 * 32];  // bf16 A tile, 4 KB
  const int tid = threadIdx.x;
  const int lane = tid & 31, w = tid >> 5;
  const int mg = w & 3, ng = w >> 2;
  const int ln15 = lane & 15, hsel = lane >> 4, sel = hsel * 8;
  const int bbase = blockIdx.x * 64;

  v8f acc[8];
#pragma unroll
  for (int t = 0; t < 8; ++t) {
    float bv = biasterm[ng * 128 + t * 16 + ln15];
#pragma unroll
    for (int j = 0; j < 8; ++j) acc[t][j] = bv;
  }
  for (int kb = 0; kb < SPOT; kb += 32) {
    __syncthreads();
    if (tid < 256) {
      int r = tid >> 2, c = (tid & 3) * 8;
      const float* src = cell + (long)(bbase + r) * SPOT + kb + c;
      *(v8bf*)&lA[r * 32 + c] = cvt8(*(const v4f*)src, *(const v4f*)(src + 4));
    }
    __syncthreads();
    v16bf a = frag16(&lA[(mg * 16 + ln15) * 32 + sel]);
    const __bf16* wb = tWbt + (long)(ng * 128 + ln15) * SPOT + kb + sel;
    v16bf f0 = frag16(wb);
    v16bf f1 = frag16(wb + 16 * SPOT);
    v16bf f2 = frag16(wb + 32 * SPOT);
    v16bf f3 = frag16(wb + 48 * SPOT);
    acc[0] = wmma_bf16(a, f0, acc[0]);
    f0 = frag16(wb + 64 * SPOT);
    acc[1] = wmma_bf16(a, f1, acc[1]);
    f1 = frag16(wb + 80 * SPOT);
    acc[2] = wmma_bf16(a, f2, acc[2]);
    f2 = frag16(wb + 96 * SPOT);
    acc[3] = wmma_bf16(a, f3, acc[3]);
    f3 = frag16(wb + 112 * SPOT);
    acc[4] = wmma_bf16(a, f0, acc[4]);
    acc[5] = wmma_bf16(a, f1, acc[5]);
    acc[6] = wmma_bf16(a, f2, acc[6]);
    acc[7] = wmma_bf16(a, f3, acc[7]);
  }
#pragma unroll
  for (int t = 0; t < 8; ++t) {
    int col = ng * 128 + t * 16 + ln15;
#pragma unroll
    for (int j = 0; j < 8; ++j) {
      int row = bbase + mg * 16 + hsel * 8 + j;
      cellterm[(long)row * SPOT + col] = acc[t][j];
    }
  }
}

// ---------------- main fused kernel -----------------------------------------
// block = (protein p, 64 batch rows); 512 threads = 16 waves.
// Stage 1: y[64][512] = pe @ combined_W^T + cellterm; double-buffered LDS with
//          TDM (tensor_load_to_lds) staging the 32KB B slab per K-step.
// Stage 2: h = leaky(y @ W1[p] + b1[p]); out = h . W2[p] + b2[p].
#define BUF_BYTES 36864  // 4 KB A tile + 32 KB B slab
#define LB0_OFF 4096
#define LB1_OFF (BUF_BYTES + 4096)
__global__ __launch_bounds__(512) void k_main(
    const float* __restrict__ pe, const __bf16* __restrict__ cWt,
    const float* __restrict__ cellterm, const __bf16* __restrict__ W1t,
    const float* __restrict__ b1, const float* __restrict__ W2,
    const float* __restrict__ b2, float* __restrict__ out) {
  extern __shared__ __align__(16) unsigned char dsm[];  // 73728 B dynamic LDS
  __bf16* lA0 = (__bf16*)dsm;
  __bf16* lB0 = (__bf16*)(dsm + LB0_OFF);
  __bf16* lA1 = (__bf16*)(dsm + BUF_BYTES);
  __bf16* lB1 = (__bf16*)(dsm + LB1_OFF);
  __bf16* yS  = (__bf16*)dsm;   // [64][512] bf16, reused after stage 1
  float* redbuf = (float*)dsm;  // [256] final reduction scratch

  const int p = blockIdx.x;
  const int bbase = blockIdx.y * 64;
  const int tid = threadIdx.x;
  const int lane = tid & 31, w = tid >> 5;
  const int mg = w & 3, ng = w >> 2;  // ng in 0..3
  const int ln15 = lane & 15, hsel = lane >> 4, sel = hsel * 8;

  const int sr = tid >> 2, sc = (tid & 3) * 8;  // sr 0..127
  const float*  peT  = pe + ((long)(bbase + (sr & 63)) * P_DIM + p) * PROT + sc;
  const __bf16* cWtT = cWt + (long)sr * PROT + sc;
  const int lAoff = (sr & 63) * 32 + sc;

  // A tile staging (fp32 -> bf16 conversion must go through registers).
  auto stageA = [&](int kb, __bf16* lAb) {
    if (tid < 256) {
      const float* src = peT + kb;
      *(v8bf*)&lAb[lAoff] = cvt8(*(const v4f*)src, *(const v4f*)(src + 4));
      __builtin_prefetch(src + 32, 0, 1);  // global_prefetch next slab
    }
  };
  // B slab staging: TDM DMA (one op per K-step, wave 0) or manual fallback.
  auto stageB = [&](int kb, __bf16* lBb, unsigned lds_off) {
#if HAVE_TDM
    if (w == 0) tdm_load_b_slab(cWt + kb, lds_off);
#else
#pragma unroll
    for (int i = 0; i < 4; ++i)
      *(v8bf*)&lBb[(i * 128 + sr) * 32 + sc] =
          *(const v8bf*)(cWtT + (long)i * 128 * PROT + kb);
#endif
  };

  // ---- stage 1 ----
  v8f acc[8];
#pragma unroll
  for (int t = 0; t < 8; ++t) {
    int col = ng * 128 + t * 16 + ln15;
#pragma unroll
    for (int j = 0; j < 8; ++j) {
      int row = bbase + mg * 16 + hsel * 8 + j;
      acc[t][j] = cellterm[(long)row * SPOT + col];   // seed = cell term + bias
    }
  }
  stageB(0, lB0, LB0_OFF);  // prologue fill of buffer 0
  stageA(0, lA0);
  int buf = 0;
  for (int kb = 0; kb < PROT; kb += 32, buf ^= 1) {
#if HAVE_TDM
    if (w == 0) __builtin_amdgcn_s_wait_tensorcnt(0);  // DMA done before barrier
#endif
    __syncthreads();  // staging for this K-step visible to all waves
    __bf16* cA = buf ? lA1 : lA0;
    __bf16* cB = buf ? lB1 : lB0;
    if (kb + 32 < PROT) {
      stageB(kb + 32, buf ? lB0 : lB1, buf ? LB0_OFF : LB1_OFF);
      stageA(kb + 32, buf ? lA0 : lA1);
    }
    const __bf16* nb = &cB[(ng * 128 + ln15) * 32 + sel];
    v16bf a = frag16(&cA[(mg * 16 + ln15) * 32 + sel]);
    v16bf f0 = frag16(nb);
    v16bf f1 = frag16(nb + 16 * 32);
#pragma unroll
    for (int t = 0; t < 8; t += 2) {
      acc[t] = wmma_bf16(a, f0, acc[t]);
      if (t + 2 < 8) f0 = frag16(nb + (t + 2) * 16 * 32);
      acc[t + 1] = wmma_bf16(a, f1, acc[t + 1]);
      if (t + 3 < 8) f1 = frag16(nb + (t + 3) * 16 * 32);
    }
  }
  __syncthreads();
#pragma unroll
  for (int t = 0; t < 8; ++t) {
    int col = ng * 128 + t * 16 + ln15;
#pragma unroll
    for (int j = 0; j < 8; ++j) {
      int row = mg * 16 + hsel * 8 + j;
      yS[row * SPOT + col] = (__bf16)acc[t][j];
    }
  }
  __syncthreads();

  // ---- stage 2 ----  wave covers 16 rows x 64 cols (4 tiles)
  v8f acc2[4];
#pragma unroll
  for (int t = 0; t < 4; ++t) {
    float bv = b1[p * HID + ng * 64 + t * 16 + ln15];
#pragma unroll
    for (int j = 0; j < 8; ++j) acc2[t][j] = bv;
  }
  for (int kb = 0; kb < SPOT; kb += 32) {
    v16bf a = frag16(&yS[(mg * 16 + ln15) * SPOT + kb + sel]);
    const __bf16* wb = W1t + ((long)p * HID + ng * 64 + ln15) * SPOT + kb + sel;
    v16bf f0 = frag16(wb);
    v16bf f1 = frag16(wb + 16 * SPOT);
    v16bf f2 = frag16(wb + 32 * SPOT);
    v16bf f3 = frag16(wb + 48 * SPOT);
    acc2[0] = wmma_bf16(a, f0, acc2[0]);
    acc2[1] = wmma_bf16(a, f1, acc2[1]);
    acc2[2] = wmma_bf16(a, f2, acc2[2]);
    acc2[3] = wmma_bf16(a, f3, acc2[3]);
  }

  // leaky-ReLU + weighted column sum (h . W2)
  float s[8];
#pragma unroll
  for (int j = 0; j < 8; ++j) s[j] = 0.f;
#pragma unroll
  for (int t = 0; t < 4; ++t) {
    float wv = W2[p * HID + ng * 64 + t * 16 + ln15];
#pragma unroll
    for (int j = 0; j < 8; ++j) {
      float h = acc2[t][j];
      h = (h >= 0.f) ? h : 0.2f * h;
      s[j] += h * wv;
    }
  }
#pragma unroll
  for (int j = 0; j < 8; ++j) {
    s[j] += __shfl_xor(s[j], 1);
    s[j] += __shfl_xor(s[j], 2);
    s[j] += __shfl_xor(s[j], 4);
    s[j] += __shfl_xor(s[j], 8);
  }
  __syncthreads();  // all yS reads done; smem reused as redbuf
  if (ln15 == 0) {  // lanes 0 and 16 carry the half-wave sums
#pragma unroll
    for (int j = 0; j < 8; ++j) {
      int row = mg * 16 + hsel * 8 + j;
      redbuf[row * 4 + ng] = s[j];
    }
  }
  __syncthreads();
  if (tid < 64) {
    float v = redbuf[tid * 4] + redbuf[tid * 4 + 1] + redbuf[tid * 4 + 2] +
              redbuf[tid * 4 + 3] + b2[p];
    out[(long)(bbase + tid) * P_DIM + p] = v;
  }
}

// ---------------- host launch ----------------------------------------------
extern "C" void kernel_launch(void* const* d_in, const int* in_sizes, int n_in,
                              void* d_out, int out_size, void* d_ws,
                              size_t ws_size, hipStream_t stream) {
  const float* cell   = (const float*)d_in[0];
  const float* pe     = (const float*)d_in[1];
  const float* projW  = (const float*)d_in[2];
  const float* projb  = (const float*)d_in[3];
  const float* transW = (const float*)d_in[4];
  const float* transb = (const float*)d_in[5];
  const float* W1     = (const float*)d_in[6];
  const float* b1     = (const float*)d_in[7];
  const float* W2     = (const float*)d_in[8];
  const float* b2     = (const float*)d_in[9];
  float* out = (float*)d_out;

  char* ws = (char*)d_ws;
  size_t off = 0;
  auto alloc = [&](size_t bytes) {
    char* pch = ws + off;
    off += (bytes + 255) & ~(size_t)255;
    return (void*)pch;
  };
  __bf16* cWt      = (__bf16*)alloc((size_t)SPOT * PROT * 2);          // 1.31 MB
  __bf16* tWbt     = (__bf16*)alloc((size_t)SPOT * SPOT * 2);          // 0.5 MB
  __bf16* W1t      = (__bf16*)alloc((size_t)P_DIM * HID * SPOT * 2);   // 16.8 MB
  float*  biasterm = (float*)alloc((size_t)SPOT * 4);
  float*  cellterm = (float*)alloc((size_t)B_DIM * SPOT * 4);          // 8.4 MB

  k_combined<<<(SPOT * PROT + 255) / 256, 256, 0, stream>>>(projW, transW, cWt);
  k_biasterm<<<(SPOT + 255) / 256, 256, 0, stream>>>(projb, transW, transb,
                                                     biasterm);
  k_tWbt<<<(SPOT * SPOT + 255) / 256, 256, 0, stream>>>(transW, tWbt);
  k_W1t<<<(P_DIM * HID * SPOT + 255) / 256, 256, 0, stream>>>(W1, W1t);
  k_cellterm<<<B_DIM / 64, 512, 0, stream>>>(cell, tWbt, biasterm, cellterm);
  dim3 grid(P_DIM, B_DIM / 64);
  k_main<<<grid, 512, 2 * BUF_BYTES, stream>>>(pe, cWt, cellterm, W1t, b1, W2,
                                               b2, out);
}
